// Agent_40063454937396
// MI455X (gfx1250) — matrix operand order repaired and checked by reading, not compiled, not run on hardware
//
#include <hip/hip_runtime.h>
#include <hip/hip_bf16.h>
#include <stdint.h>

typedef __attribute__((ext_vector_type(16))) __bf16 v16bf;
typedef __attribute__((ext_vector_type(8)))  __bf16 v8bf;
typedef __attribute__((ext_vector_type(8)))  float  v8f;

#define B_    512
#define H_    1024
#define E_    256
#define V_    8192
#define FIN_  576
#define L_    48
#define KCAT_ 1280   // E_ + H_

// ---------------- fragment loaders (per CDNA5 16-bit WMMA VGPR layouts) ---------

// A 16x32 bf16: lane holds row M=lane&15; half=lane>>4 selects K subsets
// VGPR0-3 -> K = 8*half + [0..7], VGPR4-7 -> K = 16 + 8*half + [0..7]
__device__ __forceinline__ v16bf load_a_frag(const __bf16* p) {
  v8bf lo = *reinterpret_cast<const v8bf*>(p);        // K base .. base+7
  v8bf hi = *reinterpret_cast<const v8bf*>(p + 16);   // K base+16 .. base+23
  return __builtin_shufflevector(lo, hi, 0,1,2,3,4,5,6,7,8,9,10,11,12,13,14,15);
}

// B 32x16 bf16: lane holds col N=lane&15; K = 8*half + [0..7], 16 + 8*half + [0..7]
__device__ __forceinline__ v16bf load_b_frag(const __bf16* p) {
  v8bf lo = *reinterpret_cast<const v8bf*>(p);        // K base .. base+7
  v8bf hi = *reinterpret_cast<const v8bf*>(p + 16);   // K base+16 .. base+23
  return __builtin_shufflevector(lo, hi, 0,1,2,3,4,5,6,7,8,9,10,11,12,13,14,15);
}

__device__ __forceinline__ v8f wmma_bf16(v16bf a, v16bf b, v8f c) {
  return __builtin_amdgcn_wmma_f32_16x16x32_bf16(false, a, false, b, (short)0, c,
                                                 false, false);
}

// ---------------- bf16 WMMA GEMM: C[M,N] = A[M,K] * W[N,K]^T + bias ------------
// block = 256 threads (8 waves, 4x2); wave tile 32x64; block tile 128x128.
// K unrolled by 64 with ping-pong fragment buffers (no register copies):
// each 12-load batch flies under the opposite buffer's 8 WMMAs.
// Requires K % 64 == 0 (true for 576 / 1280 / 1024).
__global__ __launch_bounds__(256)
void Agent_gemm_bf16(const __bf16* __restrict__ A, int lda,
                     const __bf16* __restrict__ W, int K,
                     const float* __restrict__ bias0,
                     const float* __restrict__ bias1,
                     float* __restrict__ C, int ldc,
                     __bf16* __restrict__ Cbf, int ldbf, int bf_ofs) {
  const int wave = threadIdx.x >> 5;
  const int lane = threadIdx.x & 31;
  const int halfId = lane >> 4;
  const int l16 = lane & 15;
  const int wm = wave >> 1;           // 0..3
  const int wn = wave & 1;            // 0..1
  const int arow0 = blockIdx.y * 128 + wm * 32 + l16;
  const int arow1 = arow0 + 16;
  const int colbase = blockIdx.x * 128 + wn * 64;

  const __bf16* pa0 = A + (size_t)arow0 * lda + 8 * halfId;
  const __bf16* pa1 = A + (size_t)arow1 * lda + 8 * halfId;
  const __bf16* pb[4];
#pragma unroll
  for (int s = 0; s < 4; ++s)
    pb[s] = W + (size_t)(colbase + s * 16 + l16) * K + 8 * halfId;

  v8f acc[8];
#pragma unroll
  for (int s = 0; s < 8; ++s)
#pragma unroll
    for (int i = 0; i < 8; ++i) acc[s][i] = 0.0f;

  // ---- ping-pong fragment buffers ----
  v16bf a0A, a1A, bA0, bA1, bA2, bA3;
  v16bf a0B, a1B, bB0, bB1, bB2, bB3;

  // prologue: chunk 0 -> buffer A
  a0A = load_a_frag(pa0);
  a1A = load_a_frag(pa1);
  bA0 = load_b_frag(pb[0]);
  bA1 = load_b_frag(pb[1]);
  bA2 = load_b_frag(pb[2]);
  bA3 = load_b_frag(pb[3]);

  for (int kc = 0; kc < K; kc += 64) {
    const int k1 = kc + 32;
    // load chunk kc+32 -> buffer B (flies under the A-buffer WMMAs)
    a0B = load_a_frag(pa0 + k1);
    a1B = load_a_frag(pa1 + k1);
    bB0 = load_b_frag(pb[0] + k1);
    bB1 = load_b_frag(pb[1] + k1);
    bB2 = load_b_frag(pb[2] + k1);
    bB3 = load_b_frag(pb[3] + k1);

    // compute chunk kc from buffer A
    acc[0] = wmma_bf16(a0A, bA0, acc[0]);
    acc[4] = wmma_bf16(a1A, bA0, acc[4]);
    acc[1] = wmma_bf16(a0A, bA1, acc[1]);
    acc[5] = wmma_bf16(a1A, bA1, acc[5]);
    acc[2] = wmma_bf16(a0A, bA2, acc[2]);
    acc[6] = wmma_bf16(a1A, bA2, acc[6]);
    acc[3] = wmma_bf16(a0A, bA3, acc[3]);
    acc[7] = wmma_bf16(a1A, bA3, acc[7]);

    // load chunk kc+64 (wraps to 0 on last iter; dead but in-bounds) -> buffer A
    const int k2 = (kc + 64 < K) ? (kc + 64) : 0;
    a0A = load_a_frag(pa0 + k2);
    a1A = load_a_frag(pa1 + k2);
    bA0 = load_b_frag(pb[0] + k2);
    bA1 = load_b_frag(pb[1] + k2);
    bA2 = load_b_frag(pb[2] + k2);
    bA3 = load_b_frag(pb[3] + k2);

    // compute chunk kc+32 from buffer B
    acc[0] = wmma_bf16(a0B, bB0, acc[0]);
    acc[4] = wmma_bf16(a1B, bB0, acc[4]);
    acc[1] = wmma_bf16(a0B, bB1, acc[1]);
    acc[5] = wmma_bf16(a1B, bB1, acc[5]);
    acc[2] = wmma_bf16(a0B, bB2, acc[2]);
    acc[6] = wmma_bf16(a1B, bB2, acc[6]);
    acc[3] = wmma_bf16(a0B, bB3, acc[3]);
    acc[7] = wmma_bf16(a1B, bB3, acc[7]);
  }

  const int mbase0 = blockIdx.y * 128 + wm * 32 + 8 * halfId;
#pragma unroll
  for (int s = 0; s < 4; ++s) {
    const int col = colbase + s * 16 + l16;
    float bsum = bias0 ? bias0[col] : 0.0f;
    if (bias1) bsum += bias1[col];
#pragma unroll
    for (int r = 0; r < 8; ++r) {
      {
        const int m = mbase0 + r;
        const float v = acc[s][r] + bsum;
        C[(size_t)m * ldc + col] = v;
        if (Cbf) Cbf[(size_t)m * ldbf + bf_ofs + col] = (__bf16)v;
      }
      {
        const int m = mbase0 + 16 + r;
        const float v = acc[4 + s][r] + bsum;
        C[(size_t)m * ldc + col] = v;
        if (Cbf) Cbf[(size_t)m * ldbf + bf_ofs + col] = (__bf16)v;
      }
    }
  }
}

// ---------------- helpers -------------------------------------------------------
__global__ void Agent_f2bf(const float* __restrict__ src, __bf16* __restrict__ dst, int n) {
  int i = blockIdx.x * blockDim.x + threadIdx.x;
  if (i < n) dst[i] = (__bf16)src[i];
}

__global__ void Agent_build_wcat(const float* __restrict__ w_ih,
                                 const float* __restrict__ w_hh,
                                 __bf16* __restrict__ wcat, int n) {
  int i = blockIdx.x * blockDim.x + threadIdx.x;
  if (i >= n) return;
  int row = i / KCAT_, k = i % KCAT_;
  float v = (k < E_) ? w_ih[(size_t)row * E_ + k] : w_hh[(size_t)row * H_ + (k - E_)];
  wcat[i] = (__bf16)v;
}

__global__ void Agent_zero_f32(float* __restrict__ p, int n) {
  int i = blockIdx.x * blockDim.x + threadIdx.x;
  if (i < n) p[i] = 0.0f;
}

__global__ void Agent_set_sos(const float* __restrict__ sos, __bf16* __restrict__ xcat, int n) {
  int i = blockIdx.x * blockDim.x + threadIdx.x;
  if (i >= n) return;
  int b = i / E_, e = i % E_;
  xcat[(size_t)b * KCAT_ + e] = (__bf16)sos[e];
}

__device__ __forceinline__ float Agent_sigmoid(float x) { return 1.0f / (1.0f + __expf(-x)); }

__global__ void Agent_lstm_cell(const float* __restrict__ gates,
                                float* __restrict__ ct,
                                __bf16* __restrict__ xcat) {
  int i = blockIdx.x * blockDim.x + threadIdx.x;   // B_*H_ threads
  int b = i >> 10, h = i & (H_ - 1);
  const float* g = gates + (size_t)b * (4 * H_);
  float ig = Agent_sigmoid(g[h]);
  float fg = Agent_sigmoid(g[H_ + h]);
  float gg = tanhf(g[2 * H_ + h]);
  float og = Agent_sigmoid(g[3 * H_ + h]);
  float c = fg * ct[i] + ig * gg;
  ct[i] = c;
  float hval = og * tanhf(c);
  xcat[(size_t)b * KCAT_ + E_ + h] = (__bf16)hval;
}

__device__ __forceinline__ uint32_t Agent_mix(uint32_t x) {
  x ^= x >> 16; x *= 0x7feb352dU; x ^= x >> 15; x *= 0x846ca68bU; x ^= x >> 16; return x;
}

// one block per batch row: softmax stats + Gumbel-max sample + embed gather
__global__ __launch_bounds__(256)
void Agent_sample(const float* __restrict__ logits,
                  const __bf16* __restrict__ emb_bf,
                  __bf16* __restrict__ xcat,
                  float* __restrict__ out, int step) {
  __shared__ float red[256];
  __shared__ int   redi[256];
  __shared__ int   s_tok;
  const int b = blockIdx.x;
  const int t = threadIdx.x;
  const float* l = logits + (size_t)b * V_;

  float m = -3.4e38f;
  for (int v = t; v < V_; v += 256) m = fmaxf(m, l[v]);
  red[t] = m; __syncthreads();
  for (int s = 128; s > 0; s >>= 1) { if (t < s) red[t] = fmaxf(red[t], red[t + s]); __syncthreads(); }
  const float mx = red[0]; __syncthreads();

  float se = 0.0f, sle = 0.0f, bestk = -3.4e38f;
  int bestv = 0;
  const uint32_t seed = Agent_mix(0x9e3779b9u ^ (uint32_t)(b * 64 + step));
  for (int v = t; v < V_; v += 256) {
    float x = l[v];
    float e = __expf(x - mx);
    se += e; sle += x * e;
    uint32_t h = Agent_mix(seed + (uint32_t)v * 0x85ebca6bu);
    float u = (float)(h >> 8) * (1.0f / 16777216.0f) + (0.5f / 16777216.0f);
    float key = x - __logf(-__logf(u));
    if (key > bestk) { bestk = key; bestv = v; }
  }
  red[t] = se; __syncthreads();
  for (int s = 128; s > 0; s >>= 1) { if (t < s) red[t] += red[t + s]; __syncthreads(); }
  const float seT = red[0]; __syncthreads();
  red[t] = sle; __syncthreads();
  for (int s = 128; s > 0; s >>= 1) { if (t < s) red[t] += red[t + s]; __syncthreads(); }
  const float sleT = red[0]; __syncthreads();
  red[t] = bestk; redi[t] = bestv; __syncthreads();
  for (int s = 128; s > 0; s >>= 1) {
    if (t < s && red[t + s] > red[t]) { red[t] = red[t + s]; redi[t] = redi[t + s]; }
    __syncthreads();
  }
  if (t == 0) {
    const int tok = redi[0];
    s_tok = tok;
    const float logZ = mx + __logf(seT);
    out[(size_t)b * L_ + step]                 = (float)tok;           // message
    out[(size_t)(B_ * L_) + b * L_ + step]     = l[tok] - logZ;        // log_prob
    out[(size_t)(2 * B_ * L_) + b * L_ + step] = logZ - sleT / seT;    // entropy
  }
  __syncthreads();
  const int tok = s_tok;
  if (t < E_) xcat[(size_t)b * KCAT_ + t] = emb_bf[(size_t)tok * E_ + t];
}

// ---------------- host launch ---------------------------------------------------
extern "C" void kernel_launch(void* const* d_in, const int* in_sizes, int n_in,
                              void* d_out, int out_size, void* d_ws, size_t ws_size,
                              hipStream_t stream) {
  const float* feat  = (const float*)d_in[0];
  const float* fc_w  = (const float*)d_in[1];
  const float* fc_b  = (const float*)d_in[2];
  const float* w_ih  = (const float*)d_in[3];
  const float* w_hh  = (const float*)d_in[4];
  const float* b_ih  = (const float*)d_in[5];
  const float* b_hh  = (const float*)d_in[6];
  const float* w_out = (const float*)d_in[7];
  const float* b_out = (const float*)d_in[8];
  const float* emb   = (const float*)d_in[9];
  const float* sos   = (const float*)d_in[10];
  float* out = (float*)d_out;

  char* ws = (char*)d_ws;
  size_t off = 0;
  auto carve = [&](size_t bytes) {
    char* p = ws + off;
    off = (off + bytes + 255) & ~(size_t)255;
    return p;
  };
  __bf16* bf_fcw  = (__bf16*)carve((size_t)H_ * FIN_ * 2);
  __bf16* bf_wcat = (__bf16*)carve((size_t)4 * H_ * KCAT_ * 2);
  __bf16* bf_wout = (__bf16*)carve((size_t)V_ * H_ * 2);
  __bf16* bf_emb  = (__bf16*)carve((size_t)V_ * E_ * 2);
  __bf16* bf_feat = (__bf16*)carve((size_t)B_ * FIN_ * 2);
  __bf16* bf_xcat = (__bf16*)carve((size_t)B_ * KCAT_ * 2);
  float*  f_ct    = (float*)carve((size_t)B_ * H_ * 4);
  float*  f_gates = (float*)carve((size_t)B_ * 4 * H_ * 4);
  float*  f_logits= (float*)carve((size_t)B_ * V_ * 4);

  const dim3 blk(256);
  auto g1 = [](int n) { return dim3((n + 255) / 256); };

  // weight / input conversions (bf16)
  Agent_f2bf<<<g1(H_ * FIN_), blk, 0, stream>>>(fc_w, bf_fcw, H_ * FIN_);
  Agent_f2bf<<<g1(V_ * H_), blk, 0, stream>>>(w_out, bf_wout, V_ * H_);
  Agent_f2bf<<<g1(V_ * E_), blk, 0, stream>>>(emb, bf_emb, V_ * E_);
  Agent_f2bf<<<g1(B_ * FIN_), blk, 0, stream>>>(feat, bf_feat, B_ * FIN_);
  Agent_build_wcat<<<g1(4 * H_ * KCAT_), blk, 0, stream>>>(w_ih, w_hh, bf_wcat, 4 * H_ * KCAT_);

  // state init
  Agent_zero_f32<<<g1(B_ * H_), blk, 0, stream>>>(f_ct, B_ * H_);
  Agent_zero_f32<<<g1(3 * B_ * L_), blk, 0, stream>>>(out, 3 * B_ * L_);
  Agent_set_sos<<<g1(B_ * E_), blk, 0, stream>>>(sos, bf_xcat, B_ * E_);

  // ht0 = feat @ fc_w^T + fc_b  (fp32 scratch -> bf16 into xcat[:,256:1280])
  Agent_gemm_bf16<<<dim3(H_ / 128, B_ / 128), blk, 0, stream>>>(
      bf_feat, FIN_, bf_fcw, FIN_, fc_b, nullptr,
      f_gates, H_, bf_xcat, KCAT_, E_);

  for (int step = 0; step < L_ - 1; ++step) {
    // gates = [et|ht] @ [w_ih|w_hh]^T + b_ih + b_hh
    Agent_gemm_bf16<<<dim3(4 * H_ / 128, B_ / 128), blk, 0, stream>>>(
        bf_xcat, KCAT_, bf_wcat, KCAT_, b_ih, b_hh,
        f_gates, 4 * H_, nullptr, 0, 0);
    Agent_lstm_cell<<<g1(B_ * H_), blk, 0, stream>>>(f_gates, f_ct, bf_xcat);
    // logits = ht @ w_out^T + b_out (A = xcat offset 256, stride 1280)
    Agent_gemm_bf16<<<dim3(V_ / 128, B_ / 128), blk, 0, stream>>>(
        bf_xcat + E_, KCAT_, bf_wout, H_, b_out, nullptr,
        f_logits, V_, nullptr, 0, 0);
    Agent_sample<<<dim3(B_), blk, 0, stream>>>(f_logits, bf_emb, bf_xcat, out, step);
  }
}
